// MutiQueryAttention_1932735283553
// MI455X (gfx1250) — compile-verified, matrix operand order checked by reading, not run on hardware
//
#include <hip/hip_runtime.h>
#include <math.h>

typedef __attribute__((ext_vector_type(16))) __bf16 v16bf;
typedef __attribute__((ext_vector_type(8)))  float  v8f;

#define HIDDEN   1024
#define HEADS    16
#define HEAD_DIM 64
#define BATCH    4
#define SEQ      2048
#define M_ROWS   (BATCH * SEQ)           // 8192
#define QKV_N    (HIDDEN + 2 * HEAD_DIM) // 1152

// ---------------------------------------------------------------------------
// Fragment loader (CDNA5 v_wmma_f32_16x16x32_bf16, ISA 7.12.2).
//   A operand from a row-major [16][>=32] tile (rows = M, contraction contig)
//   B operand from a column-major [16 cols][>=32] tile (rows = N)
// Per lane: two CONTIGUOUS 16B chunks at k = hi*8 and k = 16 + hi*8,
// row = lane&15  ->  exactly 2 x b128 loads, zero packing VALU.
// ---------------------------------------------------------------------------
static __device__ __forceinline__ v16bf frag_k32(const __bf16* base, int ld) {
  const int lane = threadIdx.x & 31;
  const int lsub = lane & 15, hi = lane >> 4;
  const __bf16* p = base + (size_t)lsub * ld + hi * 8;
  union { uint4 u[2]; v16bf v; } t;
  t.u[0] = *reinterpret_cast<const uint4*>(p);
  t.u[1] = *reinterpret_cast<const uint4*>(p + 16);
  return t.v;
}

static __device__ __forceinline__ v8f wmma_bf16(v16bf a, v16bf b, v8f c) {
  return __builtin_amdgcn_wmma_f32_16x16x32_bf16(false, a, false, b, (short)0, c,
                                                 false, false);
}

// LDS byte offset of a generic pointer to __shared__.
static __device__ __forceinline__ unsigned lds_off32(const void* p) {
  return (unsigned)(uintptr_t)p;
}

// CDNA5 async DMA: global -> LDS, 16B per lane, tracked by ASYNCcnt.
static __device__ __forceinline__ void async_copy_b128(unsigned lds, const void* sbase,
                                                       unsigned voff) {
  asm volatile("global_load_async_to_lds_b128 %0, %1, %2"
               :: "v"(lds), "v"(voff), "s"(sbase)
               : "memory");
}

// ---------------------------------------------------------------------------
// Conversion / packing kernels (one-time, tiny).
// Weights stored TRANSPOSED ([N][K] bf16) so GEMM B-fragments are wide.
// The attention scale 1/sqrt(64) = 2^-3 is folded into Wq/bq here (exact).
// ---------------------------------------------------------------------------
__global__ void convert_f32_bf16(const float* __restrict__ src,
                                 __bf16* __restrict__ dst, size_t n) {
  size_t i = (size_t)blockIdx.x * blockDim.x + threadIdx.x;
  const size_t stride = (size_t)gridDim.x * blockDim.x;
  for (; i < n; i += stride) dst[i] = (__bf16)src[i];
}

__global__ void pack_wqkv_t(const float* __restrict__ Wq, const float* __restrict__ Wk,
                            const float* __restrict__ Wv, const float* __restrict__ bq,
                            const float* __restrict__ bk, const float* __restrict__ bv,
                            __bf16* __restrict__ Wt, float* __restrict__ bqkv) {
  size_t i = (size_t)blockIdx.x * blockDim.x + threadIdx.x;
  const size_t stride = (size_t)gridDim.x * blockDim.x;
  const size_t total = (size_t)QKV_N * HIDDEN;
  for (; i < total; i += stride) {
    const int n = (int)(i / HIDDEN), k = (int)(i % HIDDEN);
    float v;
    if (n < HIDDEN)                 v = Wq[(size_t)k * HIDDEN + n] * 0.125f;  // fold 1/sqrt(Dh)
    else if (n < HIDDEN + HEAD_DIM) v = Wk[(size_t)k * HEAD_DIM + (n - HIDDEN)];
    else                            v = Wv[(size_t)k * HEAD_DIM + (n - HIDDEN - HEAD_DIM)];
    Wt[i] = (__bf16)v;  // Wt[n][k]
    if (i < QKV_N) {
      const int n2 = (int)i;
      bqkv[n2] = (n2 < HIDDEN) ? bq[n2] * 0.125f
               : (n2 < HIDDEN + HEAD_DIM) ? bk[n2 - HIDDEN]
                                          : bv[n2 - HIDDEN - HEAD_DIM];
    }
  }
}

__global__ void transpose_f32_bf16(const float* __restrict__ W,  // [K][N]
                                   __bf16* __restrict__ Wt,      // [N][K]
                                   int K, int N) {
  size_t i = (size_t)blockIdx.x * blockDim.x + threadIdx.x;
  const size_t stride = (size_t)gridDim.x * blockDim.x;
  const size_t total = (size_t)K * N;
  for (; i < total; i += stride) {
    const int n = (int)(i / K), k = (int)(i % K);
    Wt[i] = (__bf16)W[(size_t)k * N + n];
  }
}

// ---------------------------------------------------------------------------
// bf16 WMMA GEMM:  C[M,N] = A[M,K] * Wt[N,K]^T + bias[N]
// Block tile 256x128, 8 waves as 4(M) x 2(N), wave tile 64x64 (4x4 accums,
// 16 WMMAs per k-step vs 8 fragment loads). BK=32.
// Double-buffered LDS staged by global_load_async_to_lds_b128 (ASYNCcnt):
// one barrier per k-step; next tile's DMA overlaps current tile's WMMAs.
// First k-step peeled so accumulators start from WMMA's inline SRC2=0.
// ---------------------------------------------------------------------------
template <typename OutT>
__global__ __launch_bounds__(256) void gemm_bf16_wmma(
    const __bf16* __restrict__ A, const __bf16* __restrict__ Wt,
    const float* __restrict__ bias, OutT* __restrict__ C, int M, int N, int K) {
  constexpr int BM = 256, BN = 128, BK = 32;
  __shared__ __align__(16) __bf16 As[2][BM * BK];  // row-major [row][k], 2x16KB
  __shared__ __align__(16) __bf16 Bs[2][BN * BK];  // col-major [col][k], 2x8KB

  const int tid = threadIdx.x, lane = tid & 31, wave = tid >> 5;
  const int wm = wave & 3, wn = wave >> 2;  // 4 waves in M, 2 in N
  const int nb = N / BN;
  const int row0 = (blockIdx.x / nb) * BM;
  const int col0 = (blockIdx.x % nb) * BN;

  // async chunk mapping: chunk c (8 bf16) -> row c>>2, k-offset (c&3)*8
  const unsigned asBase[2] = {lds_off32(As[0]), lds_off32(As[1])};
  const unsigned bsBase[2] = {lds_off32(Bs[0]), lds_off32(Bs[1])};
  unsigned offA[4];
#pragma unroll
  for (int p = 0; p < 4; ++p) {
    const int c = tid + p * 256;
    offA[p] = (unsigned)((((c >> 2) * K) + (c & 3) * 8) * 2);
  }

  auto issue = [&](int k0, int buf) {
    const __bf16* aSrc = A + (size_t)row0 * K + k0;
    const __bf16* bSrc = Wt + (size_t)col0 * K + k0;
#pragma unroll
    for (int p = 0; p < 4; ++p)
      async_copy_b128(asBase[buf] + (unsigned)(tid + p * 256) * 16, aSrc, offA[p]);
#pragma unroll
    for (int p = 0; p < 2; ++p)
      async_copy_b128(bsBase[buf] + (unsigned)(tid + p * 256) * 16, bSrc, offA[p]);
  };

  v8f acc[4][4];
  issue(0, 0);

  // peeled first k-step: accumulate onto inline zero
  {
    asm volatile("s_wait_asynccnt 0x0" ::: "memory");
    __syncthreads();
    if (BK < K) issue(BK, 1);
    v16bf af[4], bf[4];
#pragma unroll
    for (int i = 0; i < 4; ++i) af[i] = frag_k32(&As[0][(wm * 64 + i * 16) * BK], BK);
#pragma unroll
    for (int j = 0; j < 4; ++j) bf[j] = frag_k32(&Bs[0][(wn * 64 + j * 16) * BK], BK);
    const v8f z = {};
#pragma unroll
    for (int i = 0; i < 4; ++i)
#pragma unroll
      for (int j = 0; j < 4; ++j) acc[i][j] = wmma_bf16(af[i], bf[j], z);
  }

  int buf = 1;
  for (int k0 = BK; k0 < K; k0 += BK, buf ^= 1) {
    asm volatile("s_wait_asynccnt 0x0" ::: "memory");  // our DMAs into buf landed
    __syncthreads();  // everyone's landed; everyone finished reading buf^1
    if (k0 + BK < K) issue(k0 + BK, buf ^ 1);  // DMA overlaps WMMAs below

    v16bf af[4], bf[4];
#pragma unroll
    for (int i = 0; i < 4; ++i) af[i] = frag_k32(&As[buf][(wm * 64 + i * 16) * BK], BK);
#pragma unroll
    for (int j = 0; j < 4; ++j) bf[j] = frag_k32(&Bs[buf][(wn * 64 + j * 16) * BK], BK);
#pragma unroll
    for (int i = 0; i < 4; ++i)
#pragma unroll
      for (int j = 0; j < 4; ++j) acc[i][j] = wmma_bf16(af[i], bf[j], acc[i][j]);
  }

  const int lsub = lane & 15, hi = lane >> 4;
#pragma unroll
  for (int i = 0; i < 4; ++i)
#pragma unroll
    for (int j = 0; j < 4; ++j)
#pragma unroll
      for (int r = 0; r < 8; ++r) {
        const int row = row0 + wm * 64 + i * 16 + r + 8 * hi;
        const int col = col0 + wn * 64 + j * 16 + lsub;
        C[(size_t)row * N + col] = (OutT)(acc[i][j][r] + bias[col]);
      }
}

// ---------------------------------------------------------------------------
// Multi-query flash attention, 64 keys per iteration.
// Grid: B*HEADS*(SEQ/128); block 256 = 8 waves, one 16-row Q tile each.
// Q pre-scaled by 1/sqrt(Dh) (folded into Wq). K fragments stream straight
// from global (K's [key][d] layout IS the column-major B operand; L2-resident).
// V is gathered transposed into LDS with wide stores, shared by 8 waves.
// Softmax bookkeeping (shuffle trees, m/l update, o rescale) amortized over
// 64 keys -> half the dynamic VALU/DS of a 32-key step.
// ---------------------------------------------------------------------------
__global__ __launch_bounds__(256) void mqa_attention(
    const __bf16* __restrict__ qkv, __bf16* __restrict__ attn_out) {
  __shared__ __align__(16) __bf16 Vt[HEAD_DIM * 64];  // [d][key] col-major B tile, 8KB
  __shared__ __align__(16) __bf16 Ps[8][16 * 64];     // per-wave P (row-major A tile), 16KB

  const int tid = threadIdx.x, lane = tid & 31, wave = tid >> 5;
  const int lsub = lane & 15, hi = lane >> 4;
  const int qb = blockIdx.x & 15;          // SEQ/128 = 16
  const int h  = (blockIdx.x >> 4) & 15;
  const int b  = blockIdx.x >> 8;
  const int q0 = qb * 128 + wave * 16;
  const int ld = QKV_N;

  // Q fragments (row-major A tile straight from global, 2x b128 per frag)
  const __bf16* qp = qkv + (size_t)(b * SEQ + q0) * ld + h * HEAD_DIM;
  const v16bf a_lo = frag_k32(qp, ld);
  const v16bf a_hi = frag_k32(qp + 32, ld);

  float mrow[8], lrow[8];
  v8f o[4] = {};
#pragma unroll
  for (int r = 0; r < 8; ++r) { mrow[r] = -INFINITY; lrow[r] = 0.0f; }

  const __bf16* kbase = qkv + (size_t)b * SEQ * ld + HIDDEN;
  const __bf16* vbase = kbase + HEAD_DIM;
  const int vd = tid & 63, vh = tid >> 6;  // V-transpose gather: 64 d x 4 key-quarters

  for (int kk0 = 0; kk0 < SEQ; kk0 += 64) {
    // gather V^T tile: Vt[d][key], coalesced b16 reads, 2x b128 LDS stores
    __align__(16) __bf16 tmp[16];
#pragma unroll
    for (int e = 0; e < 16; ++e)
      tmp[e] = vbase[(size_t)(kk0 + vh * 16 + e) * ld + vd];
    *reinterpret_cast<uint4*>(&Vt[vd * 64 + vh * 16])     = *reinterpret_cast<const uint4*>(&tmp[0]);
    *reinterpret_cast<uint4*>(&Vt[vd * 64 + vh * 16 + 8]) = *reinterpret_cast<const uint4*>(&tmp[8]);
    if (kk0 + 64 < SEQ)
      __builtin_prefetch(vbase + (size_t)(kk0 + 64 + vh * 16) * ld + vd, 0, 0);
    __syncthreads();

    // scores S = Q * K^T (16 x 64 keys); K frags direct from global (clause)
    v8f s[4] = {};
#pragma unroll
    for (int n = 0; n < 4; ++n) {
      const __bf16* kb = kbase + (size_t)(kk0 + n * 16) * ld;
      s[n] = wmma_bf16(a_lo, frag_k32(kb, ld), s[n]);
      s[n] = wmma_bf16(a_hi, frag_k32(kb + 32, ld), s[n]);
    }

    // online softmax over 64 keys; C-layout rows reduce over their 16-lane half
    __bf16* ps = &Ps[wave][0];
#pragma unroll
    for (int r = 0; r < 8; ++r) {
      const float x0 = s[0][r], x1 = s[1][r], x2 = s[2][r], x3 = s[3][r];
      float mx = fmaxf(fmaxf(x0, x1), fmaxf(x2, x3));
#pragma unroll
      for (int off = 8; off; off >>= 1) mx = fmaxf(mx, __shfl_xor(mx, off, 32));
      const float mnew = fmaxf(mrow[r], mx);
      const float corr = __expf(mrow[r] - mnew);
      const float p0 = __expf(x0 - mnew);
      const float p1 = __expf(x1 - mnew);
      const float p2 = __expf(x2 - mnew);
      const float p3 = __expf(x3 - mnew);
      float rs = (p0 + p1) + (p2 + p3);
#pragma unroll
      for (int off = 8; off; off >>= 1) rs += __shfl_xor(rs, off, 32);
      lrow[r] = lrow[r] * corr + rs;
      mrow[r] = mnew;
#pragma unroll
      for (int t = 0; t < 4; ++t) o[t][r] *= corr;
      const int prow = r + 8 * hi;
      ps[prow * 64 + lsub]      = (__bf16)p0;
      ps[prow * 64 + 16 + lsub] = (__bf16)p1;
      ps[prow * 64 + 32 + lsub] = (__bf16)p2;
      ps[prow * 64 + 48 + lsub] = (__bf16)p3;
    }
    asm volatile("s_wait_dscnt 0x0" ::: "memory");  // wave-private LDS RAW

    // O += P(16x64) * V(64x64): two A-frags x four 16-col B tiles
    const v16bf pf0 = frag_k32(ps, 64);
    const v16bf pf1 = frag_k32(ps + 32, 64);
#pragma unroll
    for (int t = 0; t < 4; ++t) {
      o[t] = wmma_bf16(pf0, frag_k32(&Vt[(t * 16) * 64], 64), o[t]);
      o[t] = wmma_bf16(pf1, frag_k32(&Vt[(t * 16) * 64 + 32], 64), o[t]);
    }
    __syncthreads();
  }

#pragma unroll
  for (int r = 0; r < 8; ++r) {
    const float inv = 1.0f / lrow[r];
    const int row = q0 + r + 8 * hi;
#pragma unroll
    for (int t = 0; t < 4; ++t) {
      const int col = h * HEAD_DIM + t * 16 + lsub;
      attn_out[(size_t)(b * SEQ + row) * HIDDEN + col] = (__bf16)(o[t][r] * inv);
    }
  }
}

// ---------------------------------------------------------------------------
// Host-side launch
// ---------------------------------------------------------------------------
extern "C" void kernel_launch(void* const* d_in, const int* in_sizes, int n_in,
                              void* d_out, int out_size, void* d_ws, size_t ws_size,
                              hipStream_t stream) {
  (void)in_sizes; (void)n_in; (void)out_size; (void)ws_size;
  const float* X  = (const float*)d_in[0];
  const float* Wq = (const float*)d_in[1];
  const float* bq = (const float*)d_in[2];
  const float* Wk = (const float*)d_in[3];
  const float* bk = (const float*)d_in[4];
  const float* Wv = (const float*)d_in[5];
  const float* bv = (const float*)d_in[6];
  const float* Wo = (const float*)d_in[7];
  const float* bo = (const float*)d_in[8];
  float* out = (float*)d_out;

  char* ws = (char*)d_ws;
  size_t off = 0;
  auto take = [&](size_t bytes) {
    void* p = ws + off;
    off += (bytes + 255) & ~(size_t)255;
    return p;
  };
  __bf16* Xbf   = (__bf16*)take((size_t)M_ROWS * HIDDEN * 2);  // 16 MB
  __bf16* WqkvT = (__bf16*)take((size_t)QKV_N * HIDDEN * 2);   // 2.25 MB  [N][K]
  __bf16* WoT   = (__bf16*)take((size_t)HIDDEN * HIDDEN * 2);  // 2 MB     [N][K]
  float*  bqkv  = (float*)take((size_t)QKV_N * 4);
  __bf16* QKV   = (__bf16*)take((size_t)M_ROWS * QKV_N * 2);   // 18 MB
  __bf16* Attn  = (__bf16*)take((size_t)M_ROWS * HIDDEN * 2);  // 16 MB

  // 1) precision conversion + transposed weight packing (one-time)
  convert_f32_bf16<<<1024, 256, 0, stream>>>(X, Xbf, (size_t)M_ROWS * HIDDEN);
  pack_wqkv_t<<<512, 256, 0, stream>>>(Wq, Wk, Wv, bq, bk, bv, WqkvT, bqkv);
  transpose_f32_bf16<<<512, 256, 0, stream>>>(Wo, WoT, HIDDEN, HIDDEN);

  // 2) fused QKV projection: [8192,1024] x [1024,1152] + bias -> bf16
  gemm_bf16_wmma<__bf16><<<(M_ROWS / 256) * (QKV_N / 128), 256, 0, stream>>>(
      Xbf, WqkvT, bqkv, QKV, M_ROWS, QKV_N, HIDDEN);

  // 3) flash multi-query attention -> bf16 [8192,1024]
  mqa_attention<<<BATCH * HEADS * (SEQ / 128), 256, 0, stream>>>(QKV, Attn);

  // 4) output projection: [8192,1024] x [1024,1024] + bo -> f32 out
  gemm_bf16_wmma<float><<<(M_ROWS / 256) * (HIDDEN / 128), 256, 0, stream>>>(
      Attn, WoT, bo, out, M_ROWS, HIDDEN, HIDDEN);
}
